// GraphNetwork_61289183314447
// MI455X (gfx1250) — matrix operand order repaired and checked by reading, not compiled, not run on hardware
//
#include <hip/hip_runtime.h>
#include <hip/hip_bf16.h>

#define NODES   50000
#define EDGES   400000
#define F_IN    128
#define GC      441
#define GCP     448      // GC padded to multiple of 32 for WMMA K
#define FC1D    1024
#define FC2D    1024
#define ACTD    512
#define KSTEPS_MAX (GCP / 32)   // 14

typedef __attribute__((ext_vector_type(16))) __bf16 v16bf;
typedef __attribute__((ext_vector_type(8)))  __bf16 v8bf;
typedef __attribute__((ext_vector_type(8)))  float  v8f;

static __device__ __forceinline__ __bf16 f2bf(float f) {
    union { float f; unsigned u; } x; x.f = f;
    unsigned r = (x.u + 0x7FFFu + ((x.u >> 16) & 1u)) >> 16;   // round-to-nearest-even
    unsigned short h = (unsigned short)r;
    __bf16 out;
    __builtin_memcpy(&out, &h, 2);
    return out;
}

// ---------------- degree / normalization ----------------
__global__ void k_deg_init(float* deg) {
    int i = blockIdx.x * blockDim.x + threadIdx.x;
    if (i < NODES) deg[i] = 1.0f;                 // self-loop
}
__global__ void k_deg_count(const long long* __restrict__ dst, float* deg) {
    int e = blockIdx.x * blockDim.x + threadIdx.x;
    if (e < EDGES) atomicAdd(&deg[(int)dst[e]], 1.0f);
}
__global__ void k_dinv(const float* __restrict__ deg, float* dinv) {
    int i = blockIdx.x * blockDim.x + threadIdx.x;
    if (i < NODES) dinv[i] = rsqrtf(deg[i]);
}

// ---------------- fp32 -> bf16 (with column zero-pad) ----------------
__global__ void k_f32_to_bf16(const float* __restrict__ src, __bf16* __restrict__ dst,
                              int rows, int cols, int dstStride) {
    size_t idx = (size_t)blockIdx.x * blockDim.x + threadIdx.x;
    size_t total = (size_t)rows * dstStride;
    if (idx >= total) return;
    int c = (int)(idx % dstStride);
    int r = (int)(idx / dstStride);
    float v = (c < cols) ? src[(size_t)r * cols + c] : 0.0f;
    dst[idx] = f2bf(v);
}

// relu(agg + bias) -> bf16 padded (feeds GEMM2)
__global__ void k_bias_relu_bf16(const float* __restrict__ agg, const float* __restrict__ bias,
                                 __bf16* __restrict__ dst) {
    size_t idx = (size_t)blockIdx.x * blockDim.x + threadIdx.x;
    size_t total = (size_t)NODES * GCP;
    if (idx >= total) return;
    int c = (int)(idx % GCP);
    int r = (int)(idx / GCP);
    float v = 0.0f;
    if (c < GC) v = fmaxf(agg[(size_t)r * GC + c] + bias[c], 0.0f);
    dst[idx] = f2bf(v);
}

// ---------------- WMMA GEMM: Out[M x 441] = A[M x Ks](bf16) * W[Kw x 441](f32) ----------------
// block = 256 threads = 8 waves. Each wave computes a 32x32 output tile (2M x 2N subtiles,
// 4 accumulators, 4 WMMA per K-step). Weight tile pre-swizzled in LDS into B-fragment
// layout so each lane reads its whole fragment as one contiguous 32B chunk (2x ds_load_b128).
__global__ __launch_bounds__(256)
void k_gemm_bf16(const __bf16* __restrict__ A, int lda,
                 const float* __restrict__ W, int Kw,
                 float* __restrict__ Out, int M, int Ks) {
    // [step][half][lane][16] bf16 : 14*2*32*16*2B = 28672 B
    __shared__ __bf16 sW[KSTEPS_MAX * 2 * 32 * 16];

    const int lane = threadIdx.x & 31;
    const int wid  = threadIdx.x >> 5;
    const int n0   = blockIdx.x * 32;
    const int nsteps = Ks >> 5;

    // Stage + swizzle weight tile. Fragment layout for step s, half h:
    //   lane l = (n&15) + ((k&8)?16:0),  element j = (k&7) + ((k&16)?8:0),  n = h*16 + (n&15)
    for (int idx = threadIdx.x; idx < Ks * 32; idx += 256) {
        int k = idx >> 5;
        int c = idx & 31;
        int gn = n0 + c;
        float v = (k < Kw && gn < GC) ? W[(size_t)k * GC + gn] : 0.0f;
        int s = k >> 5;
        int h = c >> 4;
        int l = (c & 15) + ((k & 8) ? 16 : 0);
        int j = (k & 7) + ((k & 16) ? 8 : 0);
        sW[((((s << 1) + h) << 5) + l) * 16 + j] = f2bf(v);
    }
    __syncthreads();

    const int tileM0 = (blockIdx.y * 8 + wid) * 2;
    if (tileM0 * 16 >= M) return;                 // whole wave out of range
    const int tileM1 = tileM0 + 1;
    const bool m1ok = (tileM1 * 16) < M;          // M is a multiple of 16

    const int mr     = lane & 15;
    const int kbaseA = (lane < 16) ? 0 : 8;       // 16-bit A 16x32 lane striping
    const __bf16* Arow0 = A + (size_t)(tileM0 * 16 + mr) * lda;
    const __bf16* Arow1 = A + (size_t)((m1ok ? tileM1 : tileM0) * 16 + mr) * lda;

    v8f acc00 = {}, acc01 = {}, acc10 = {}, acc11 = {};
    for (int s = 0; s < nsteps; ++s) {
        const int k0 = s << 5;
        __builtin_prefetch(Arow0 + k0 + 64, 0, 1);
        __builtin_prefetch(Arow1 + k0 + 64, 0, 1);

        // A fragments: elements 0..7 -> K=k0+kbaseA+0..7 ; 8..15 -> +16..23
        v8bf a0lo = *(const v8bf*)(Arow0 + k0 + kbaseA);
        v8bf a0hi = *(const v8bf*)(Arow0 + k0 + kbaseA + 16);
        v8bf a1lo = *(const v8bf*)(Arow1 + k0 + kbaseA);
        v8bf a1hi = *(const v8bf*)(Arow1 + k0 + kbaseA + 16);
        v16bf a0, a1;
#pragma unroll
        for (int i = 0; i < 8; ++i) {
            a0[i] = a0lo[i]; a0[i + 8] = a0hi[i];
            a1[i] = a1lo[i]; a1[i + 8] = a1hi[i];
        }

        // B fragments: contiguous 32B per lane (pre-swizzled)
        const __bf16* bp = sW + (((s << 1) << 5) + lane) * 16;
        v16bf b0 = *(const v16bf*)bp;
        v16bf b1 = *(const v16bf*)(bp + 32 * 16);

        acc00 = __builtin_amdgcn_wmma_f32_16x16x32_bf16(false, a0, false, b0, (short)0, acc00, false, false);
        acc01 = __builtin_amdgcn_wmma_f32_16x16x32_bf16(false, a0, false, b1, (short)0, acc01, false, false);
        acc10 = __builtin_amdgcn_wmma_f32_16x16x32_bf16(false, a1, false, b0, (short)0, acc10, false, false);
        acc11 = __builtin_amdgcn_wmma_f32_16x16x32_bf16(false, a1, false, b1, (short)0, acc11, false, false);
    }

    // C/D layout: VGPR r -> row base + r (+8 for lanes 16..31), col = lane&15 within subtile
    const int r0  = (lane < 16) ? 0 : 8;
    const int gn0 = n0 + (lane & 15);
    const int gn1 = gn0 + 16;
    {
        const int gm = tileM0 * 16 + r0;
        if (gn0 < GC) {
#pragma unroll
            for (int r = 0; r < 8; ++r) Out[(size_t)(gm + r) * GC + gn0] = acc00[r];
        }
        if (gn1 < GC) {
#pragma unroll
            for (int r = 0; r < 8; ++r) Out[(size_t)(gm + r) * GC + gn1] = acc01[r];
        }
    }
    if (m1ok) {
        const int gm = tileM1 * 16 + r0;
        if (gn0 < GC) {
#pragma unroll
            for (int r = 0; r < 8; ++r) Out[(size_t)(gm + r) * GC + gn0] = acc10[r];
        }
        if (gn1 < GC) {
#pragma unroll
            for (int r = 0; r < 8; ++r) Out[(size_t)(gm + r) * GC + gn1] = acc11[r];
        }
    }
}

// ---------------- conv1 aggregation ----------------
__global__ void k_selfloop(const float* __restrict__ h, const float* __restrict__ dinv,
                           float* __restrict__ agg) {
    size_t idx = (size_t)blockIdx.x * blockDim.x + threadIdx.x;
    size_t total = (size_t)NODES * GC;
    if (idx >= total) return;
    int i = (int)(idx / GC);
    float di = dinv[i];
    agg[idx] = h[idx] * di * di;
}

__global__ __launch_bounds__(256)
void k_edge_scatter(const long long* __restrict__ src, const long long* __restrict__ dst,
                    const float* __restrict__ dinv, const float* __restrict__ h,
                    float* __restrict__ agg) {
    int e = blockIdx.x * 8 + (threadIdx.x >> 5);  // one wave per edge
    if (e >= EDGES) return;
    int lane = threadIdx.x & 31;
    int s = (int)src[e], d = (int)dst[e];
    float norm = dinv[s] * dinv[d];
    const float* hr = h + (size_t)s * GC;
    float* ar = agg + (size_t)d * GC;
    for (int f = lane; f < GC; f += 32)
        atomicAdd(&ar[f], norm * hr[f]);
}

// ---------------- conv2 @ node 0 only (exact: final output uses row 0 only) ----------------
__global__ void k_node0_init(const float* __restrict__ h2, const float* __restrict__ dinv,
                             float* __restrict__ tmp) {
    int f = blockIdx.x * blockDim.x + threadIdx.x;
    if (f < GC) {
        float d0 = dinv[0];
        tmp[f] = h2[f] * d0 * d0;
    }
}
__global__ __launch_bounds__(256)
void k_node0_edges(const long long* __restrict__ src, const long long* __restrict__ dst,
                   const float* __restrict__ dinv, const float* __restrict__ h2,
                   float* __restrict__ tmp) {
    int e = blockIdx.x * 8 + (threadIdx.x >> 5);
    if (e >= EDGES) return;
    if ((int)dst[e] != 0) return;                 // only edges into node 0
    int lane = threadIdx.x & 31;
    int s = (int)src[e];
    float norm = dinv[s] * dinv[0];
    const float* hr = h2 + (size_t)s * GC;
    for (int f = lane; f < GC; f += 32)
        atomicAdd(&tmp[f], norm * hr[f]);
}
__global__ void k_node0_relu(const float* __restrict__ tmp, const float* __restrict__ bias,
                             float* __restrict__ v) {
    int f = blockIdx.x * blockDim.x + threadIdx.x;
    if (f < GC) v[f] = fmaxf(tmp[f] + bias[f], 0.0f);
}

// ---------------- small FC layers (GEMV, coalesced over output dim) ----------------
__global__ void k_fc_relu(const float* __restrict__ in, const float* __restrict__ W,
                          const float* __restrict__ b, float* __restrict__ out,
                          int IN, int OUT) {
    int j = blockIdx.x * blockDim.x + threadIdx.x;
    if (j >= OUT) return;
    float acc = b[j];
    for (int i = 0; i < IN; ++i)
        acc = fmaf(in[i], W[(size_t)i * OUT + j], acc);
    out[j] = fmaxf(acc, 0.0f);
}

// ---------------- host launcher ----------------
extern "C" void kernel_launch(void* const* d_in, const int* in_sizes, int n_in,
                              void* d_out, int out_size, void* d_ws, size_t ws_size,
                              hipStream_t stream) {
    const float*     state = (const float*)d_in[0];
    const long long* ei    = (const long long*)d_in[1];   // int64 [2, E]
    const long long* esrc  = ei;
    const long long* edst  = ei + EDGES;
    const float* w_gc1 = (const float*)d_in[2];
    const float* b_gc1 = (const float*)d_in[3];
    const float* w_gc2 = (const float*)d_in[4];
    const float* b_gc2 = (const float*)d_in[5];
    const float* w_fc1 = (const float*)d_in[6];
    const float* b_fc1 = (const float*)d_in[7];
    const float* w_fc2 = (const float*)d_in[8];
    const float* b_fc2 = (const float*)d_in[9];
    const float* w_fc3 = (const float*)d_in[10];
    const float* b_fc3 = (const float*)d_in[11];
    float* out = (float*)d_out;

    char* ws = (char*)d_ws;
    size_t off = 0;
    auto alloc = [&](size_t bytes) -> void* {
        void* p = ws + off;
        off += (bytes + 255) & ~(size_t)255;
        return p;
    };
    float*  deg    = (float*)alloc(sizeof(float) * NODES);
    float*  dinv   = (float*)alloc(sizeof(float) * NODES);
    __bf16* stateb = (__bf16*)alloc(sizeof(__bf16) * (size_t)NODES * F_IN);
    float*  h      = (float*)alloc(sizeof(float) * (size_t)NODES * GC);   // h1 then h2
    float*  agg    = (float*)alloc(sizeof(float) * (size_t)NODES * GC);
    __bf16* x1     = (__bf16*)alloc(sizeof(__bf16) * (size_t)NODES * GCP);
    float*  tmp441 = (float*)alloc(sizeof(float) * GCP);
    float*  v441   = (float*)alloc(sizeof(float) * GCP);
    float*  f1     = (float*)alloc(sizeof(float) * FC1D);
    float*  f2     = (float*)alloc(sizeof(float) * FC2D);
    (void)ws_size; (void)in_sizes; (void)n_in; (void)out_size;

    // 1) symmetric-normalization coefficients
    k_deg_init <<<(NODES + 255) / 256, 256, 0, stream>>>(deg);
    k_deg_count<<<(EDGES + 255) / 256, 256, 0, stream>>>(edst, deg);
    k_dinv     <<<(NODES + 255) / 256, 256, 0, stream>>>(deg, dinv);

    const int mtilesPerBlk = 16;                   // 8 waves x 2 M-tiles
    dim3 ggrid((GC + 31) / 32, (NODES / 16 + mtilesPerBlk - 1) / mtilesPerBlk);  // 14 x 196

    // 2) GEMM1: h1 = bf16(state) @ bf16(W_gc1), f32 accumulate
    {
        size_t tot = (size_t)NODES * F_IN;
        k_f32_to_bf16<<<(unsigned)((tot + 255) / 256), 256, 0, stream>>>(
            state, stateb, NODES, F_IN, F_IN);
        k_gemm_bf16<<<ggrid, 256, 0, stream>>>(stateb, F_IN, w_gc1, F_IN, h, NODES, F_IN);
    }

    // 3) conv1 aggregation: agg = D^-1/2 (A+I) D^-1/2 h1
    {
        size_t tot = (size_t)NODES * GC;
        k_selfloop<<<(unsigned)((tot + 255) / 256), 256, 0, stream>>>(h, dinv, agg);
        k_edge_scatter<<<EDGES / 8, 256, 0, stream>>>(esrc, edst, dinv, h, agg);
    }

    // 4) x1 = bf16(relu(agg + b_gc1)), K zero-padded 441 -> 448
    {
        size_t tot = (size_t)NODES * GCP;
        k_bias_relu_bf16<<<(unsigned)((tot + 255) / 256), 256, 0, stream>>>(agg, b_gc1, x1);
    }

    // 5) GEMM2: h2 = x1 @ bf16(W_gc2)
    k_gemm_bf16<<<ggrid, 256, 0, stream>>>(x1, GCP, w_gc2, GC, h, NODES, GCP);

    // 6) conv2 aggregation at node 0 only, then relu(+bias)
    k_node0_init <<<(GC + 255) / 256, 256, 0, stream>>>(h, dinv, tmp441);
    k_node0_edges<<<EDGES / 8, 256, 0, stream>>>(esrc, edst, dinv, h, tmp441);
    k_node0_relu <<<(GC + 255) / 256, 256, 0, stream>>>(tmp441, b_gc2, v441);

    // 7) FC head
    k_fc_relu<<<(FC1D + 255) / 256, 256, 0, stream>>>(v441, w_fc1, b_fc1, f1, GC, FC1D);
    k_fc_relu<<<(FC2D + 255) / 256, 256, 0, stream>>>(f1, w_fc2, b_fc2, f2, FC1D, FC2D);
    k_fc_relu<<<(ACTD + 255) / 256, 256, 0, stream>>>(f2, w_fc3, b_fc3, out, FC2D, ACTD);
}
